// SelfAttention_6511170421561
// MI455X (gfx1250) — matrix operand order repaired and checked by reading, not compiled, not run on hardware
//
#include <hip/hip_runtime.h>

// ---------------------------------------------------------------------------
// Self-attention (1x1-conv QKV + softmax(QK^T)V) for MI455X / gfx1250.
//   Pass 1: fp32 -> bf16 QKV projection (memory-light, scalar).
//   Pass 2: flash attention, v_wmma_f32_16x16x32_bf16 with fp32 accumulate.
// Dataflow: per block (4 waves, 64 queries) K/V key-tiles are streamed
// global -> LDS with CDNA5 async loads (global_load_async_to_lds_b128,
// ASYNCcnt), double-buffered so tile t+1 loads while tile t feeds the WMMAs.
// ---------------------------------------------------------------------------

typedef __bf16 bf16;
typedef __attribute__((ext_vector_type(16))) __bf16 bf16x16;
typedef __attribute__((ext_vector_type(8)))  __bf16 bf16x8;
typedef __attribute__((ext_vector_type(8)))  float  f32x8;

static constexpr int Bb  = 4;
static constexpr int Cc  = 128;
static constexpr int CQ  = 64;
static constexpr int Nn  = 64 * 64;           // 4096 spatial positions
static constexpr float L2E = 1.44269504088896340736f;

// LDS tile geometry: rows padded to 72 bf16 (144 B = 9*16 B) to avoid bank
// conflicts on the 16-lane strided b128 fragment reads.
static constexpr int LDP    = 72;
static constexpr int KBUF_E = 64  * LDP;       // K tile: 64 keys x 64 ch
static constexpr int VBUF_E = 128 * LDP;       // V tile: 128 ch  x 64 keys
static constexpr int PT_E   = 16  * LDP;       // P tile per wave: 16 x 64

// ---------------------------------------------------------------------------
__device__ __forceinline__ void async_b128(const bf16* g, const bf16* l)
{
    // LDS aperture: addr[31:0] is the LDS byte address (ISA 10.2).
    unsigned lofs = (unsigned)(uintptr_t)l;
    asm volatile("global_load_async_to_lds_b128 %0, %1, off"
                 :: "v"(lofs), "v"(g) : "memory");
}
__device__ __forceinline__ void wait_async0()
{
    asm volatile("s_wait_asynccnt 0x0" ::: "memory");
}

// ---------------------------------------------------------------------------
// Kernel 1: QKV projection (1x1 convs == channel matmuls), fp32 in, bf16 out.
//   q,k stored [B][N][CQ] row-major  -> A/B fragments of S-GEMM contiguous
//   v   stored [B][C][N]  row-major  -> B fragments of O-GEMM contiguous
// ---------------------------------------------------------------------------
__global__ __launch_bounds__(256)
void qkv_proj_kernel(const float* __restrict__ x,
                     const float* __restrict__ Wq, const float* __restrict__ bq,
                     const float* __restrict__ Wk, const float* __restrict__ bk,
                     const float* __restrict__ Wv, const float* __restrict__ bv,
                     bf16* __restrict__ q, bf16* __restrict__ k,
                     bf16* __restrict__ v)
{
    __shared__ float xsh[Cc * 64];            // x[b, :, n0:n0+64] tile (32 KB)
    const int tile = blockIdx.x;              // Bb * (Nn/64) tiles
    const int b    = tile / (Nn / 64);
    const int n0   = (tile % (Nn / 64)) * 64;
    const int t    = threadIdx.x;

    for (int idx = t; idx < Cc * 64; idx += 256) {
        const int c  = idx >> 6;
        const int nn = idx & 63;
        xsh[idx] = x[((size_t)b * Cc + c) * Nn + n0 + nn];
    }
    __syncthreads();

    const float* Wrow;
    float bias;
    int kind, o;
    if (t < 64)       { kind = 0; o = t;       Wrow = Wq + o * Cc; bias = bq[o]; }
    else if (t < 128) { kind = 1; o = t - 64;  Wrow = Wk + o * Cc; bias = bk[o]; }
    else              { kind = 2; o = t - 128; Wrow = Wv + o * Cc; bias = bv[o]; }

    for (int chunk = 0; chunk < 4; ++chunk) {
        float acc[16];
        #pragma unroll
        for (int i = 0; i < 16; ++i) acc[i] = bias;
        for (int c = 0; c < Cc; ++c) {
            const float w = Wrow[c];
            #pragma unroll
            for (int i = 0; i < 16; ++i)
                acc[i] += w * xsh[c * 64 + chunk * 16 + i];
        }
        #pragma unroll
        for (int i = 0; i < 16; ++i) {
            const int n = n0 + chunk * 16 + i;
            const bf16 h = (bf16)acc[i];
            if (kind == 0)      q[((size_t)b * Nn + n) * CQ + o] = h;
            else if (kind == 1) k[((size_t)b * Nn + n) * CQ + o] = h;
            else                v[((size_t)b * Cc + o) * Nn + n] = h;
        }
    }
}

// ---------------------------------------------------------------------------
// Kernel 2: flash attention. 4 waves / block share async-staged K/V tiles.
// ---------------------------------------------------------------------------
__global__ __launch_bounds__(128)
void flash_attn_kernel(const bf16* __restrict__ q, const bf16* __restrict__ k,
                       const bf16* __restrict__ v, float* __restrict__ out)
{
    // [ kbuf x2 | vbuf x2 | ptile x4 ]  = 64512 bytes
    __shared__ alignas(16) bf16 smem[2 * KBUF_E + 2 * VBUF_E + 4 * PT_E];

    const int lane = threadIdx.x & 31;
    const int wv   = threadIdx.x >> 5;
    const int half = lane >> 4;
    const int l16  = lane & 15;

    const int b  = blockIdx.x / (Nn / 64);    // batch
    const int i0 = (blockIdx.x % (Nn / 64)) * 64 + wv * 16;   // query tile

    const bf16* kb = k + (size_t)b * Nn * CQ;
    const bf16* vb = v + (size_t)b * Cc * Nn;
    bf16* pt = smem + 2 * KBUF_E + 2 * VBUF_E + wv * PT_E;

    // ---- stage key tile j0 into LDS buffer `dst` (cooperative, async) ----
    auto stage = [&](int j0, int dst) {
        bf16* kd = smem + dst * KBUF_E;
        bf16* vd = smem + 2 * KBUF_E + dst * VBUF_E;
        #pragma unroll
        for (int u = 0; u < 4; ++u) {         // K tile: 64 rows x 8 x 16B
            const int idx = threadIdx.x + u * 128;
            const int row = idx >> 3, cc = idx & 7;
            async_b128(kb + (size_t)(j0 + row) * CQ + cc * 8,
                       kd + row * LDP + cc * 8);
        }
        #pragma unroll
        for (int u = 0; u < 8; ++u) {         // V tile: 128 rows x 8 x 16B
            const int idx = threadIdx.x + u * 128;
            const int row = idx >> 3, cc = idx & 7;
            async_b128(vb + (size_t)row * Nn + j0 + cc * 8,
                       vd + row * LDP + cc * 8);
        }
    };

    // ---- load Q A-fragments once (16 rows x 64 ch -> two 16x32 frags).
    // A-frag k order used here: lane L row M=L&15, elem i -> K=(L>>4)*16+i (same as B)
    bf16x16 aq[2];
    {
        const bf16* qrow = q + ((size_t)b * Nn + i0 + l16) * CQ;
        #pragma unroll
        for (int kc = 0; kc < 2; ++kc) {
            bf16x8 lo = *(const bf16x8*)(qrow + kc * 32 + half * 16);
            bf16x8 hi = *(const bf16x8*)(qrow + kc * 32 + half * 16 + 8);
            aq[kc] = __builtin_shufflevector(lo, hi, 0, 1, 2, 3, 4, 5, 6, 7,
                                                     8, 9, 10, 11, 12, 13, 14, 15);
        }
    }

    f32x8 fzero;
    #pragma unroll
    for (int r = 0; r < 8; ++r) fzero[r] = 0.0f;
    f32x8 accO[8];
    #pragma unroll
    for (int ct = 0; ct < 8; ++ct) accO[ct] = fzero;
    float mrow[8], lrow[8];                   // row r -> query i0 + r + 8*half
    #pragma unroll
    for (int r = 0; r < 8; ++r) { mrow[r] = -1.0e30f; lrow[r] = 0.0f; }

    // ---- prologue: stage first tile ----
    stage(0, 0);
    wait_async0();
    __syncthreads();

    constexpr int NT = Nn / 64;               // 64 key tiles of 64
    for (int it = 0; it < NT; ++it) {
        const int buf = it & 1;
        if (it + 1 < NT) stage((it + 1) * 64, buf ^ 1);   // overlap next tile

        const bf16* kcur = smem + buf * KBUF_E;
        const bf16* vcur = smem + 2 * KBUF_E + buf * VBUF_E;

        // ---- S = Q K^T for 64 keys (4 D-tiles), B-frags from LDS.
        // B-frag k order used here (matches A): lane L col N=L&15, elem i -> K=(L>>4)*16+i.
        f32x8 S[4];
        #pragma unroll
        for (int jt = 0; jt < 4; ++jt) S[jt] = fzero;
        #pragma unroll
        for (int kc = 0; kc < 2; ++kc) {
            #pragma unroll
            for (int jt = 0; jt < 4; ++jt) {
                bf16x16 bk = *(const bf16x16*)(kcur + (jt * 16 + l16) * LDP +
                                               kc * 32 + half * 16);
                S[jt] = __builtin_amdgcn_wmma_f32_16x16x32_bf16(
                            false, aq[kc], false, bk, (short)0, S[jt],
                            false, false);
            }
        }

        // ---- online softmax across 64 columns (one reduction per tile).
        float alpha[8];
        #pragma unroll
        for (int r = 0; r < 8; ++r) {
            float mx = fmaxf(fmaxf(S[0][r], S[1][r]), fmaxf(S[2][r], S[3][r]));
            #pragma unroll
            for (int m = 1; m <= 8; m <<= 1)
                mx = fmaxf(mx, __shfl_xor(mx, m, 32));
            const float nm = fmaxf(mrow[r], mx);
            alpha[r] = exp2f((mrow[r] - nm) * L2E);
            mrow[r]  = nm;
            float rs = 0.0f;
            #pragma unroll
            for (int jt = 0; jt < 4; ++jt) {
                const float p = exp2f((S[jt][r] - nm) * L2E);
                S[jt][r] = p;
                rs += p;
            }
            #pragma unroll
            for (int m = 1; m <= 8; m <<= 1)
                rs += __shfl_xor(rs, m, 32);
            lrow[r] = lrow[r] * alpha[r] + rs;
        }

        // ---- P: C/D layout -> LDS -> A layout (16x64 bf16, per-wave region).
        #pragma unroll
        for (int jt = 0; jt < 4; ++jt) {
            #pragma unroll
            for (int r = 0; r < 8; ++r)
                pt[(r + 8 * half) * LDP + jt * 16 + l16] = (bf16)S[jt][r];
        }
        __syncthreads();

        // ---- O(16x128) += P(16x64) * V^T(64x16): two K=32 sub-steps.
        #pragma unroll
        for (int jj = 0; jj < 2; ++jj) {
            bf16x8 plo = *(const bf16x8*)(pt + l16 * LDP + jj * 32 + half * 16);
            bf16x8 phi = *(const bf16x8*)(pt + l16 * LDP + jj * 32 + half * 16 + 8);
            bf16x16 pa = __builtin_shufflevector(plo, phi,
                             0, 1, 2, 3, 4, 5, 6, 7,
                             8, 9, 10, 11, 12, 13, 14, 15);
            #pragma unroll
            for (int ct = 0; ct < 8; ++ct) {
                bf16x16 bvf = *(const bf16x16*)(vcur + (ct * 16 + l16) * LDP +
                                                jj * 32 + half * 16);
                f32x8 cin;
                if (jj == 0) {
                    #pragma unroll
                    for (int r = 0; r < 8; ++r) cin[r] = accO[ct][r] * alpha[r];
                } else {
                    cin = accO[ct];
                }
                accO[ct] = __builtin_amdgcn_wmma_f32_16x16x32_bf16(
                               false, pa, false, bvf, (short)0, cin,
                               false, false);
            }
        }

        wait_async0();                        // tile t+1 landed in LDS
        __syncthreads();                      // all waves done with tile t
    }

    // ---- normalize and store: out[b][c][i], fp32.
    #pragma unroll
    for (int r = 0; r < 8; ++r) lrow[r] = 1.0f / lrow[r];
    #pragma unroll
    for (int ct = 0; ct < 8; ++ct) {
        const int c = ct * 16 + l16;
        #pragma unroll
        for (int r = 0; r < 8; ++r) {
            const int i = i0 + r + 8 * half;
            out[((size_t)b * Cc + c) * Nn + i] = accO[ct][r] * lrow[r];
        }
    }
}

// ---------------------------------------------------------------------------
extern "C" void kernel_launch(void* const* d_in, const int* in_sizes, int n_in,
                              void* d_out, int out_size, void* d_ws,
                              size_t ws_size, hipStream_t stream)
{
    const float* x  = (const float*)d_in[0];
    const float* Wq = (const float*)d_in[1];
    const float* bq = (const float*)d_in[2];
    const float* Wk = (const float*)d_in[3];
    const float* bk = (const float*)d_in[4];
    const float* Wv = (const float*)d_in[5];
    const float* bv = (const float*)d_in[6];
    float* out = (float*)d_out;

    // Workspace: q(2MB) | k(2MB) | v(4MB) bf16, 8 MB total.
    bf16* qws = (bf16*)d_ws;
    bf16* kws = qws + (size_t)Bb * Nn * CQ;
    bf16* vws = kws + (size_t)Bb * Nn * CQ;

    qkv_proj_kernel<<<Bb * (Nn / 64), 256, 0, stream>>>(
        x, Wq, bq, Wk, bk, Wv, bv, qws, kws, vws);

    flash_attn_kernel<<<Bb * (Nn / 64), 128, 0, stream>>>(
        qws, kws, vws, out);
}
